// model001_32117765439634
// MI455X (gfx1250) — matrix operand order repaired and checked by reading, not compiled, not run on hardware
//
#include <hip/hip_runtime.h>

// ---------------------------------------------------------------------------
// Linear RNN: x_{t+1} = x_t@Wx + u_t@Wu + b ; Y[t] = x_{t+1}@Wx2y + b
// Split-bf16 (hi+lo) WMMA path: 3x v_wmma_f32_16x16x32_bf16 per 32-K chunk,
// fp32 accumulation -> ~fp32 accuracy at bf16 WMMA throughput.
// One kernel launch per recurrence step (latency-bound; Wx lives in L2).
// ---------------------------------------------------------------------------

typedef __attribute__((ext_vector_type(16))) __bf16 bf16x16;
typedef __attribute__((ext_vector_type(8)))  __bf16 bf16x8;
typedef __attribute__((ext_vector_type(8)))  float  f32x8;
typedef __attribute__((ext_vector_type(4)))  float  f32x4;

static constexpr int kB  = 256;   // batch
static constexpr int kNY = 64;
static constexpr int kNU = 64;
static constexpr int kNH = 1024;
static constexpr int kT  = 512;

// ---- fragment loaders (ISA 7.12.2 layouts, wave32) -------------------------

// A 16x32 bf16 tile, row-major source. Lane holds row (lane&15);
// v[0..7] = A[row][k0 + 8*kh .. +8], v[8..15] = A[row][k0+16+8*kh .. +8].
__device__ __forceinline__ bf16x16 load_frag_a(const __bf16* A, int lda,
                                               int row, int k0, int kh) {
  const __bf16* p = A + (size_t)row * lda + k0 + 8 * kh;
  bf16x8 a0 = *(const bf16x8*)p;
  bf16x8 a1 = *(const bf16x8*)(p + 16);
  bf16x16 r;
#pragma unroll
  for (int i = 0; i < 8; ++i) { r[i] = a0[i]; r[i + 8] = a1[i]; }
  return r;
}

// B 32x16 bf16 tile from a pre-TRANSPOSED weight (BT[n][k], ldb == K).
// Lane holds column (lane&15); v[0..7] = B[k0+8*kh .. +8][col], v[8..15] = B[k0+16+8*kh .. +8][col].
__device__ __forceinline__ bf16x16 load_frag_b(const __bf16* BT, int ldb,
                                               int col, int k0, int kh) {
  const __bf16* p = BT + (size_t)col * ldb + k0 + 8 * kh;
  bf16x8 b0 = *(const bf16x8*)p;
  bf16x8 b1 = *(const bf16x8*)(p + 16);
  bf16x16 r;
#pragma unroll
  for (int i = 0; i < 8; ++i) { r[i] = b0[i]; r[i + 8] = b1[i]; }
  return r;
}

// A tile from fp32 source, split on the fly into hi/lo bf16 fragments.
__device__ __forceinline__ void load_frag_a_f32(const float* A, int lda,
                                                int row, int k0, int kh,
                                                bf16x16& hi, bf16x16& lo) {
  const float* p = A + (size_t)row * lda + k0 + 8 * kh;
  f32x4 v0 = *(const f32x4*)p;
  f32x4 v1 = *(const f32x4*)(p + 4);
  f32x4 v2 = *(const f32x4*)(p + 16);
  f32x4 v3 = *(const f32x4*)(p + 20);
  float v[16];
#pragma unroll
  for (int i = 0; i < 4; ++i) {
    v[i] = v0[i]; v[4 + i] = v1[i]; v[8 + i] = v2[i]; v[12 + i] = v3[i];
  }
#pragma unroll
  for (int i = 0; i < 16; ++i) {
    __bf16 h = (__bf16)v[i];
    hi[i] = h;
    lo[i] = (__bf16)(v[i] - (float)h);
  }
}

__device__ __forceinline__ f32x8 wmma_bf16(bf16x16 a, bf16x16 b, f32x8 c) {
  return __builtin_amdgcn_wmma_f32_16x16x32_bf16(
      /*neg_a=*/false, a, /*neg_b=*/false, b,
      /*c_mod=*/(short)0, c, /*reuse_a=*/false, /*reuse_b=*/false);
}

// acc += A * B with split-bf16 A (stored hi/lo) and split-bf16 B^T (hi/lo).
__device__ __forceinline__ void gemm_tile_split(
    const __bf16* Ahi, const __bf16* Alo, int lda,
    const __bf16* BThi, const __bf16* BTlo, int ldb,
    int K, int m0, int n0, int lane, f32x8& acc) {
  const int kh = lane >> 4, lr = lane & 15;
  for (int kc = 0; kc < K; kc += 32) {
    bf16x16 ah = load_frag_a(Ahi, lda, m0 + lr, kc, kh);
    bf16x16 al = load_frag_a(Alo, lda, m0 + lr, kc, kh);
    bf16x16 bh = load_frag_b(BThi, ldb, n0 + lr, kc, kh);
    bf16x16 bl = load_frag_b(BTlo, ldb, n0 + lr, kc, kh);
    acc = wmma_bf16(ah, bh, acc);
    acc = wmma_bf16(ah, bl, acc);
    acc = wmma_bf16(al, bh, acc);
  }
}

// acc += A * B with fp32 A (split on the fly) and split-bf16 B^T.
__device__ __forceinline__ void gemm_tile_f32a(
    const float* A, int lda,
    const __bf16* BThi, const __bf16* BTlo, int ldb,
    int K, int m0, int n0, int lane, f32x8& acc) {
  const int kh = lane >> 4, lr = lane & 15;
  for (int kc = 0; kc < K; kc += 32) {
    bf16x16 ah, al;
    load_frag_a_f32(A, lda, m0 + lr, kc, kh, ah, al);
    bf16x16 bh = load_frag_b(BThi, ldb, n0 + lr, kc, kh);
    bf16x16 bl = load_frag_b(BTlo, ldb, n0 + lr, kc, kh);
    acc = wmma_bf16(ah, bh, acc);
    acc = wmma_bf16(ah, bl, acc);
    acc = wmma_bf16(al, bh, acc);
  }
}

// Store a 16x16 f32 accumulator tile as a hi/lo bf16 state pair.
__device__ __forceinline__ void store_tile_split(f32x8 acc, __bf16* Xhi, __bf16* Xlo,
                                                 int ld, int m0, int n0, int lane) {
  const int kh = lane >> 4, lr = lane & 15;
#pragma unroll
  for (int r = 0; r < 8; ++r) {
    const int row = m0 + r + 8 * kh;
    const int col = n0 + lr;
    float v = acc[r];
    __bf16 h = (__bf16)v;
    Xhi[(size_t)row * ld + col] = h;
    Xlo[(size_t)row * ld + col] = (__bf16)(v - (float)h);
  }
}

// y-projection tile: y = x @ Wx2y + bx2y, fp32 output.
__device__ __forceinline__ void y_tile(const __bf16* xhi, const __bf16* xlo,
                                       const __bf16* Wx2yT_hi, const __bf16* Wx2yT_lo,
                                       const float* bx2y, float* yout,
                                       int tile, int lane) {
  const int kh = lane >> 4, lr = lane & 15;
  const int m0 = (tile >> 2) * 16;  // 16 m-tiles over batch
  const int n0 = (tile & 3) * 16;   // 4 n-tiles over NY=64
  f32x8 acc;
  const float b = bx2y[n0 + lr];
#pragma unroll
  for (int r = 0; r < 8; ++r) acc[r] = b;
  gemm_tile_split(xhi, xlo, kNH, Wx2yT_hi, Wx2yT_lo, kNH, kNH, m0, n0, lane, acc);
#pragma unroll
  for (int r = 0; r < 8; ++r)
    yout[(size_t)(m0 + r + 8 * kh) * kNY + n0 + lr] = acc[r];
}

// ---- kernels ---------------------------------------------------------------

// Transpose + hi/lo-split every weight matrix once.
__global__ void prep_kernel(const float* __restrict__ Wy2x,
                            const float* __restrict__ Wxu2x,
                            const float* __restrict__ Wx2y,
                            __bf16* WxT_hi, __bf16* WxT_lo,
                            __bf16* WuT_hi, __bf16* WuT_lo,
                            __bf16* Wy2xT_hi, __bf16* Wy2xT_lo,
                            __bf16* Wx2yT_hi, __bf16* Wx2yT_lo) {
  const int gid = blockIdx.x * blockDim.x + threadIdx.x;
  if (gid < kNH * kNH) {  // WxT[n][k] = Wxu2x[k][n]
    const int n = gid / kNH, k = gid % kNH;
    const float a = Wxu2x[(size_t)k * kNH + n];
    const __bf16 h = (__bf16)a;
    WxT_hi[gid] = h; WxT_lo[gid] = (__bf16)(a - (float)h);
  }
  if (gid < kNH * kNU) {
    {  // WuT[n][k] = Wxu2x[NH+k][n]   (n<NH, k<NU)
      const int n = gid / kNU, k = gid % kNU;
      const float a = Wxu2x[(size_t)(kNH + k) * kNH + n];
      const __bf16 h = (__bf16)a;
      WuT_hi[gid] = h; WuT_lo[gid] = (__bf16)(a - (float)h);
    }
    {  // Wy2xT[n][k] = Wy2x[k][n]     (n<NH, k<NY)
      const int n = gid / kNY, k = gid % kNY;
      const float a = Wy2x[(size_t)k * kNH + n];
      const __bf16 h = (__bf16)a;
      Wy2xT_hi[gid] = h; Wy2xT_lo[gid] = (__bf16)(a - (float)h);
    }
    {  // Wx2yT[n][k] = Wx2y[k][n]     (n<NY, k<NH)
      const int n = gid / kNH, k = gid % kNH;
      const float a = Wx2y[(size_t)k * kNY + n];
      const __bf16 h = (__bf16)a;
      Wx2yT_hi[gid] = h; Wx2yT_lo[gid] = (__bf16)(a - (float)h);
    }
  }
}

// x0 = y0 @ Wy2x + by2x  -> split state (1024 tile-waves)
__global__ void x0_kernel(const float* __restrict__ y0,
                          const __bf16* __restrict__ Wy2xT_hi,
                          const __bf16* __restrict__ Wy2xT_lo,
                          const float* __restrict__ by2x,
                          __bf16* xhi, __bf16* xlo) {
  const int wave = blockIdx.x * (blockDim.x >> 5) + (threadIdx.x >> 5);
  const int lane = threadIdx.x & 31;
  const int lr = lane & 15;
  const int m0 = (wave >> 6) * 16;  // 16 m-tiles (batch)
  const int n0 = (wave & 63) * 16;  // 64 n-tiles (NH)
  f32x8 acc;
  const float b = by2x[n0 + lr];
#pragma unroll
  for (int r = 0; r < 8; ++r) acc[r] = b;
  gemm_tile_f32a(y0, kNY, Wy2xT_hi, Wy2xT_lo, kNY, kNY, m0, n0, lane, acc);
  store_tile_split(acc, xhi, xlo, kNH, m0, n0, lane);
}

// One recurrence step: waves 0..1023 compute x_out = x_in@Wx + u@Wu + b;
// waves 1024..1087 compute y = x_in@Wx2y + bx2y (deferred-by-one output).
__global__ void step_kernel(const __bf16* __restrict__ xin_hi,
                            const __bf16* __restrict__ xin_lo,
                            __bf16* xout_hi, __bf16* xout_lo,
                            const float* __restrict__ Ustep,
                            const __bf16* __restrict__ WxT_hi,
                            const __bf16* __restrict__ WxT_lo,
                            const __bf16* __restrict__ WuT_hi,
                            const __bf16* __restrict__ WuT_lo,
                            const float* __restrict__ bxu2x,
                            const __bf16* __restrict__ Wx2yT_hi,
                            const __bf16* __restrict__ Wx2yT_lo,
                            const float* __restrict__ bx2y,
                            float* yout) {
  const int wave = blockIdx.x * (blockDim.x >> 5) + (threadIdx.x >> 5);
  const int lane = threadIdx.x & 31;
  if (wave < 1024) {
    const int lr = lane & 15;
    const int m0 = (wave >> 6) * 16;  // batch tiles
    const int n0 = (wave & 63) * 16;  // NH tiles
    f32x8 acc;
    const float b = bxu2x[n0 + lr];
#pragma unroll
    for (int r = 0; r < 8; ++r) acc[r] = b;
    gemm_tile_split(xin_hi, xin_lo, kNH, WxT_hi, WxT_lo, kNH, kNH, m0, n0, lane, acc);
    gemm_tile_f32a(Ustep, kNU, WuT_hi, WuT_lo, kNU, kNU, m0, n0, lane, acc);
    store_tile_split(acc, xout_hi, xout_lo, kNH, m0, n0, lane);
  } else if (yout != nullptr) {
    y_tile(xin_hi, xin_lo, Wx2yT_hi, Wx2yT_lo, bx2y, yout, wave - 1024, lane);
  }
}

// Final output row: y_511 from x_512 (64 tile-waves).
__global__ void ylast_kernel(const __bf16* __restrict__ xhi,
                             const __bf16* __restrict__ xlo,
                             const __bf16* __restrict__ Wx2yT_hi,
                             const __bf16* __restrict__ Wx2yT_lo,
                             const float* __restrict__ bx2y,
                             float* yout) {
  const int wave = blockIdx.x * (blockDim.x >> 5) + (threadIdx.x >> 5);
  const int lane = threadIdx.x & 31;
  y_tile(xhi, xlo, Wx2yT_hi, Wx2yT_lo, bx2y, yout, wave, lane);
}

// ---- host ------------------------------------------------------------------

extern "C" void kernel_launch(void* const* d_in, const int* in_sizes, int n_in,
                              void* d_out, int out_size, void* d_ws, size_t ws_size,
                              hipStream_t stream) {
  (void)in_sizes; (void)n_in; (void)out_size; (void)ws_size;
  const float* y0    = (const float*)d_in[0];
  const float* U     = (const float*)d_in[1];
  const float* Wy2x  = (const float*)d_in[2];
  const float* by2x  = (const float*)d_in[3];
  const float* Wxu2x = (const float*)d_in[4];
  const float* bxu2x = (const float*)d_in[5];
  const float* Wx2y  = (const float*)d_in[6];
  const float* bx2y  = (const float*)d_in[7];
  float* Y = (float*)d_out;

  char* ws = (char*)d_ws;
  size_t off = 0;
  auto take = [&](size_t elems) -> __bf16* {
    __bf16* p = (__bf16*)(ws + off);
    off += elems * sizeof(__bf16);
    return p;
  };
  __bf16* WxT_hi   = take((size_t)kNH * kNH);
  __bf16* WxT_lo   = take((size_t)kNH * kNH);
  __bf16* WuT_hi   = take((size_t)kNH * kNU);
  __bf16* WuT_lo   = take((size_t)kNH * kNU);
  __bf16* Wy2xT_hi = take((size_t)kNH * kNY);
  __bf16* Wy2xT_lo = take((size_t)kNH * kNY);
  __bf16* Wx2yT_hi = take((size_t)kNY * kNH);
  __bf16* Wx2yT_lo = take((size_t)kNY * kNH);
  __bf16* xh[2], *xl[2];
  xh[0] = take((size_t)kB * kNH); xl[0] = take((size_t)kB * kNH);
  xh[1] = take((size_t)kB * kNH); xl[1] = take((size_t)kB * kNH);

  prep_kernel<<<(kNH * kNH + 255) / 256, 256, 0, stream>>>(
      Wy2x, Wxu2x, Wx2y, WxT_hi, WxT_lo, WuT_hi, WuT_lo,
      Wy2xT_hi, Wy2xT_lo, Wx2yT_hi, Wx2yT_lo);

  // 1024 tiles, 8 waves/block -> 128 blocks
  x0_kernel<<<128, 256, 0, stream>>>(y0, Wy2xT_hi, Wy2xT_lo, by2x, xh[0], xl[0]);

  // 1024 x-tiles + 64 y-tiles = 1088 waves -> 136 blocks
  for (int s = 0; s < kT; ++s) {
    const int pi = s & 1, po = pi ^ 1;
    float* yout = (s >= 1) ? (Y + (size_t)(s - 1) * kB * kNY) : nullptr;
    step_kernel<<<136, 256, 0, stream>>>(
        xh[pi], xl[pi], xh[po], xl[po],
        U + (size_t)s * kB * kNU,
        WxT_hi, WxT_lo, WuT_hi, WuT_lo, bxu2x,
        Wx2yT_hi, Wx2yT_lo, bx2y, yout);
  }

  // y_511 from x_512 (which lives in buffer 0 after 512 steps)
  ylast_kernel<<<8, 256, 0, stream>>>(xh[0], xl[0], Wx2yT_hi, Wx2yT_lo, bx2y,
                                      Y + (size_t)(kT - 1) * kB * kNY);
}
